// knn_GradCAM_21002390078208
// MI455X (gfx1250) — matrix operand order repaired and checked by reading, not compiled, not run on hardware
//
#include <hip/hip_runtime.h>
#include <stdint.h>

typedef __attribute__((ext_vector_type(16))) __bf16 v16bf;
typedef __attribute__((ext_vector_type(8)))  __bf16 v8bf;
typedef __attribute__((ext_vector_type(8)))  float  v8f;

#define CN      25000
#define CE      200000
#define CN2     50000
#define CA      100000
#define CE2     400000
#define CG      200
#define CF_IN   40
#define CB_IN   10
#define CDIM    64
#define CNI2    16
#define CHID    128

#define KF0     5184   /* pad(40*128 + 40, 32) */
#define KF1     4128   /* 32*128 + 32 (already multiple of 32) */

// ---------------------------------------------------------------------------
// elementwise / packing kernels
// ---------------------------------------------------------------------------

__global__ void k_zero_f32(float* __restrict__ p, long long n) {
  long long t = (long long)blockIdx.x * blockDim.x + threadIdx.x;
  if (t < n) p[t] = 0.0f;
}

// fp32 [M,K] row-major -> bf16 [M,Kpad] row-major, zero padded
__global__ void k_cvt_pad_bf16(const float* __restrict__ src, __bf16* __restrict__ dst,
                               int M, int K, int Kpad) {
  long long t = (long long)blockIdx.x * blockDim.x + threadIdx.x;
  long long total = (long long)M * Kpad;
  if (t >= total) return;
  int m = (int)(t / Kpad);
  int k = (int)(t - (long long)m * Kpad);
  dst[t] = (k < K) ? (__bf16)src[(long long)m * K + k] : (__bf16)0.0f;
}

// fp32 B [K,N] row-major -> bf16 packed B^T [N][Kpad] (zero pad K)
__global__ void k_pack_B(const float* __restrict__ B, __bf16* __restrict__ dst,
                         int K, int Ncols, int Kpad) {
  long long t = (long long)blockIdx.x * blockDim.x + threadIdx.x;
  long long total = (long long)Ncols * Kpad;
  if (t >= total) return;
  int n = (int)(t / Kpad);
  int k = (int)(t - (long long)n * Kpad);
  dst[t] = (k < K) ? (__bf16)B[(long long)k * Ncols + n] : (__bf16)0.0f;
}

// Repack NNConv edge-MLP second-layer weights for the fused outer-product GEMM.
// Flat K index f:  f < m_in*128      : i = f>>7, k = f&127 -> W2[k][i*m_out+o]
//                  m_in*128 <= f     : t = f - m_in*128; t<m_in -> b2[t*m_out+o]
// dst layout: [m_out][Kfpad] (B^T packed), bf16.
__global__ void k_pack_W2(const float* __restrict__ W2, const float* __restrict__ b2,
                          __bf16* __restrict__ dst, int m_in, int m_out, int Kfpad) {
  long long t = (long long)blockIdx.x * blockDim.x + threadIdx.x;
  long long total = (long long)m_out * Kfpad;
  if (t >= total) return;
  int o = (int)(t / Kfpad);
  int f = (int)(t - (long long)o * Kfpad);
  float v = 0.0f;
  int mflat = m_in << 7;
  if (f < mflat) {
    int i = f >> 7, k = f & 127;
    v = W2[(long long)k * (m_in * m_out) + i * m_out + o];
  } else {
    int i = f - mflat;
    if (i < m_in) v = b2[i * m_out + o];
  }
  dst[t] = (__bf16)v;
}

// ---------------------------------------------------------------------------
// generic wave32 WMMA GEMM:  out = act( A[M,Kpad](bf16) @ Bp[N][Kpad]^T + add + bias )
// one wave per block, 16 rows per block, NT 16-wide column tiles.
// ---------------------------------------------------------------------------
template<int NT>
__global__ __launch_bounds__(32)
void k_gemm_wmma(const __bf16* __restrict__ A, int lda,
                 const __bf16* __restrict__ Bp, int ldb,
                 const float* __restrict__ bias,
                 const float* __restrict__ addsrc,
                 float* __restrict__ outf,
                 __bf16* __restrict__ outb, int ldob,
                 int M, int Ncols, int Kpad, int relu) {
  const int lane = threadIdx.x;
  const bool hi  = lane >= 16;
  const int l16  = lane & 15;
  const long long m0 = (long long)blockIdx.x * 16;

  long long mload = m0 + l16;
  if (mload >= M) mload = M - 1;
  const __bf16* arow = A + mload * lda;

  v8f zero = {};
  v8f acc[NT];
#pragma unroll
  for (int t = 0; t < NT; ++t) acc[t] = zero;

  const int kch = Kpad >> 5;
  for (int c = 0; c < kch; ++c) {
    const int kb = (c << 5) + (hi ? 8 : 0);
    v8bf alo = *(const v8bf*)(arow + kb);
    v8bf ahi = *(const v8bf*)(arow + kb + 16);
    v16bf a;
#pragma unroll
    for (int j = 0; j < 8; ++j) { a[j] = alo[j]; a[j + 8] = ahi[j]; }
#pragma unroll
    for (int t = 0; t < NT; ++t) {
      const __bf16* brow = Bp + (long long)(t * 16 + l16) * ldb + (c << 5) + (hi ? 8 : 0);
      v8bf blo = *(const v8bf*)brow;
      v8bf bhi = *(const v8bf*)(brow + 16);
      v16bf b;
#pragma unroll
      for (int j = 0; j < 8; ++j) { b[j] = blo[j]; b[j + 8] = bhi[j]; }
      acc[t] = __builtin_amdgcn_wmma_f32_16x16x32_bf16(false, a, false, b,
                                                       (short)0, acc[t], false, false);
    }
  }

#pragma unroll
  for (int t = 0; t < NT; ++t) {
    int n = t * 16 + l16;
#pragma unroll
    for (int r = 0; r < 8; ++r) {
      long long m = m0 + r + (hi ? 8 : 0);
      if (m < M) {
        float v = acc[t][r];
        if (addsrc) v += addsrc[m * Ncols + n];
        if (bias)   v += bias[n];
        if (relu)   v = v > 0.0f ? v : 0.0f;
        if (outf) outf[m * Ncols + n] = v;
        if (outb) outb[m * ldob + n] = (__bf16)v;
      }
    }
  }
}

// ---------------------------------------------------------------------------
// Fused NNConv message kernel.
//   msg[e,o] = sum_{i,k} x[src[e],i] * h[e,k] * W2r[i*128+k, o]   (+ b2 tail block)
// A-matrix fragments P[e, i*128+k] = x*h are generated on the fly from LDS tiles
// using packed bf16 multiplies (v_pk_mul_bf16) so A-gen co-executes cheaply with
// the WMMA pipe.  Results scatter-added to agg[dst[e]] with f32 atomics.
// MT*16 edges per block, one wave per block.
// ---------------------------------------------------------------------------
template<int MT, int NT>
__global__ __launch_bounds__(32)
void k_edge_fused(const __bf16* __restrict__ xb, int ldx, int m_in,
                  const __bf16* __restrict__ hh,           // [E][128] bf16
                  const __bf16* __restrict__ Bp, int ldb,  // [NT*16][Kfpad]
                  const int* __restrict__ src, const int* __restrict__ dst,
                  float* __restrict__ agg,                 // [N][NT*16] f32
                  long long E, int Kfpad) {
  const int EPB = MT * 16;
  __shared__ __align__(16) __bf16 hl[MT * 16 * 128];
  __shared__ __align__(16) __bf16 xl[MT * 16 * 64];
  __shared__ int sl[MT * 16];
  __shared__ int dl[MT * 16];

  const int lane = threadIdx.x;
  const bool hi  = lane >= 16;
  const int l16  = lane & 15;
  const long long e0 = (long long)blockIdx.x * EPB;
  const int cnt_e = (int)((E - e0 < EPB) ? (E - e0) : EPB);

  for (int m = lane; m < EPB; m += 32) {
    long long e = e0 + m;
    int ee = (int)((e < E) ? e : (E - 1));
    sl[m] = src[ee];
    dl[m] = dst[ee];
  }
  __syncthreads();

  // gather x rows (bf16, row stride ldx elements) into LDS (row stride 64 el)
  const int KXd = ldx >> 1;  // dwords per row
  for (int t = lane; t < EPB * KXd; t += 32) {
    int m = t / KXd;
    int q = t - m * KXd;
    ((unsigned int*)xl)[m * 32 + q] =
        ((const unsigned int*)(xb + (long long)sl[m] * ldx))[q];
  }
  // copy contiguous h tile: cnt_e * 128 bf16 = cnt_e*16 uint4
  {
    const uint4* hg = (const uint4*)(hh + e0 * 128);
    uint4* hls = (uint4*)hl;
    for (int t = lane; t < cnt_e * 16; t += 32) hls[t] = hg[t];
  }
  __syncthreads();

  v8f zero = {};
  v8f acc[MT][NT];
#pragma unroll
  for (int mt = 0; mt < MT; ++mt)
#pragma unroll
    for (int t = 0; t < NT; ++t) acc[mt][t] = zero;

  const int kch = Kfpad >> 5;
  const int mainch = m_in << 2;  // chunks covering the outer-product region

  for (int c = 0; c < kch; ++c) {
    v16bf bfrag[NT];
#pragma unroll
    for (int t = 0; t < NT; ++t) {
      const __bf16* brow = Bp + (long long)(t * 16 + l16) * ldb + (c << 5) + (hi ? 8 : 0);
      v8bf blo = *(const v8bf*)brow;
      v8bf bhi = *(const v8bf*)(brow + 16);
#pragma unroll
      for (int j = 0; j < 8; ++j) { bfrag[t][j] = blo[j]; bfrag[t][j + 8] = bhi[j]; }
    }
#pragma unroll
    for (int mt = 0; mt < MT; ++mt) {
      const int mrow = mt * 16 + l16;
      v16bf a;
      if (c < mainch) {
        const int i  = c >> 2;
        const int k0 = (c & 3) << 5;
        const __bf16 xsb = xl[mrow * 64 + i];
        v8bf xsv;
#pragma unroll
        for (int j = 0; j < 8; ++j) xsv[j] = xsb;
        const int kb = k0 + (hi ? 8 : 0);
        const __bf16* hrow = hl + mrow * 128;
        v8bf lo = *(const v8bf*)(hrow + kb);
        v8bf hg = *(const v8bf*)(hrow + kb + 16);
        v8bf plo = lo * xsv;   // packed bf16 multiplies
        v8bf phi = hg * xsv;
#pragma unroll
        for (int j = 0; j < 8; ++j) { a[j] = plo[j]; a[j + 8] = phi[j]; }
      } else {
        // bias tail block: P[e, m_in*128 + i] = x[e,i]
        const int base = ((c - mainch) << 5) + (hi ? 8 : 0);
#pragma unroll
        for (int j = 0; j < 8; ++j) {
          int i1 = base + j, i2 = base + 16 + j;
          a[j]     = (i1 < m_in) ? xl[mrow * 64 + i1] : (__bf16)0.0f;
          a[j + 8] = (i2 < m_in) ? xl[mrow * 64 + i2] : (__bf16)0.0f;
        }
      }
#pragma unroll
      for (int t = 0; t < NT; ++t)
        acc[mt][t] = __builtin_amdgcn_wmma_f32_16x16x32_bf16(false, a, false, bfrag[t],
                                                             (short)0, acc[mt][t],
                                                             false, false);
    }
  }

  const int Ncols = NT * 16;
#pragma unroll
  for (int mt = 0; mt < MT; ++mt)
#pragma unroll
    for (int t = 0; t < NT; ++t)
#pragma unroll
      for (int r = 0; r < 8; ++r) {
        int m = mt * 16 + r + (hi ? 8 : 0);
        if (e0 + m < E) {
          long long node = dl[m];
          atomicAdd(&agg[node * Ncols + (t * 16 + l16)], acc[mt][t][r]);
        }
      }
}

// ---------------------------------------------------------------------------
// scatter / pooling kernels
// ---------------------------------------------------------------------------

// outsum[sidx[r]] += vals[gidx?gidx[r]:r];  cnt[sidx[r]] += 1 (once per row)
__global__ void k_scatter_add(const float* __restrict__ vals, const int* __restrict__ gidx,
                              const int* __restrict__ sidx, float* __restrict__ outsum,
                              float* __restrict__ cnt, long long rows, int F) {
  long long t = (long long)blockIdx.x * blockDim.x + threadIdx.x;
  if (t >= rows * F) return;
  int r = (int)(t / F);
  int f = (int)(t - (long long)r * F);
  long long srow = gidx ? (long long)gidx[r] : (long long)r;
  atomicAdd(&outsum[(long long)sidx[r] * F + f], vals[srow * F + f]);
  if (f == 0 && cnt) atomicAdd(&cnt[sidx[r]], 1.0f);
}

// h2 = [mean(sum2/cnt2) | iso] -> f32 [N2,80] and bf16 [N2,96] (zero padded)
__global__ void k_finalize_h2(const float* __restrict__ sum, const float* __restrict__ cnt,
                              const float* __restrict__ iso,
                              float* __restrict__ h2f, __bf16* __restrict__ h2b) {
  long long t = (long long)blockIdx.x * blockDim.x + threadIdx.x;
  if (t >= (long long)CN2 * 96) return;
  int n = (int)(t / 96);
  int k = (int)(t - (long long)n * 96);
  float v = 0.0f;
  if (k < CDIM) {
    float c = cnt[n]; c = c > 1.0f ? c : 1.0f;
    v = sum[(long long)n * CDIM + k] / c;
  } else if (k < CDIM + CNI2) {
    v = iso[(long long)n * CNI2 + (k - CDIM)];
  }
  if (k < CDIM + CNI2) h2f[(long long)n * 80 + k] = v;
  h2b[t] = (__bf16)v;
}

__global__ void k_finalize_emb(const float* __restrict__ sum, const float* __restrict__ cnt,
                               float* __restrict__ emb, int off) {
  int t = blockIdx.x * blockDim.x + threadIdx.x;
  if (t >= CG * CDIM) return;
  int g = t / CDIM, k = t - g * CDIM;
  float c = cnt[g]; c = c > 1.0f ? c : 1.0f;
  emb[g * 2 * CDIM + off + k] = sum[t] / c;
}

__global__ void k_final_mlp(const float* __restrict__ emb,
                            const float* __restrict__ W1, const float* __restrict__ b1,
                            const float* __restrict__ W2, const float* __restrict__ b2,
                            const float* __restrict__ W3, const float* __restrict__ b3,
                            float* __restrict__ out) {
  int g = blockIdx.x * blockDim.x + threadIdx.x;
  if (g >= CG) return;
  const float* e = emb + g * 2 * CDIM;
  float h1[CDIM];
  for (int o = 0; o < CDIM; ++o) {
    float s = b1[o];
#pragma unroll 8
    for (int i = 0; i < 2 * CDIM; ++i) s += e[i] * W1[i * CDIM + o];
    h1[o] = s > 0.0f ? s : 0.0f;
  }
  float h2[CDIM / 2];
  for (int o = 0; o < CDIM / 2; ++o) {
    float s = b2[o];
#pragma unroll 8
    for (int i = 0; i < CDIM; ++i) s += h1[i] * W2[i * (CDIM / 2) + o];
    h2[o] = s > 0.0f ? s : 0.0f;
  }
  float s = b3[0];
#pragma unroll 8
  for (int i = 0; i < CDIM / 2; ++i) s += h2[i] * W3[i];
  out[g] = s;
}

// ---------------------------------------------------------------------------
// launcher
// ---------------------------------------------------------------------------

static inline size_t aln256(size_t x) { return (x + 255) & ~(size_t)255; }
static inline int gridFor(long long n, int b) { return (int)((n + b - 1) / b); }

extern "C" void kernel_launch(void* const* d_in, const int* in_sizes, int n_in,
                              void* d_out, int out_size, void* d_ws, size_t ws_size,
                              hipStream_t stream) {
  (void)in_sizes; (void)n_in; (void)out_size;
  const float* x         = (const float*)d_in[0];
  const float* edge_attr = (const float*)d_in[1];
  const float* iso2      = (const float*)d_in[2];
  const int*   eidx      = (const int*)d_in[3];
  const int*   assign    = (const int*)d_in[4];
  const int*   eidx2     = (const int*)d_in[5];
  const int*   batch     = (const int*)d_in[6];
  const int*   batch2    = (const int*)d_in[7];
  const float* nn0_W1 = (const float*)d_in[8];
  const float* nn0_b1 = (const float*)d_in[9];
  const float* nn0_W2 = (const float*)d_in[10];
  const float* nn0_b2 = (const float*)d_in[11];
  const float* root0  = (const float*)d_in[12];
  const float* bias0  = (const float*)d_in[13];
  const float* nn1_W1 = (const float*)d_in[14];
  const float* nn1_b1 = (const float*)d_in[15];
  const float* nn1_W2 = (const float*)d_in[16];
  const float* nn1_b2 = (const float*)d_in[17];
  const float* root1  = (const float*)d_in[18];
  const float* bias1  = (const float*)d_in[19];
  const float* W4rel  = (const float*)d_in[20];
  const float* W4root = (const float*)d_in[21];
  const float* b4     = (const float*)d_in[22];
  const float* W5rel  = (const float*)d_in[23];
  const float* W5root = (const float*)d_in[24];
  const float* b5     = (const float*)d_in[25];
  const float* fc1_W  = (const float*)d_in[26];
  const float* fc1_b  = (const float*)d_in[27];
  const float* fc2_W  = (const float*)d_in[28];
  const float* fc2_b  = (const float*)d_in[29];
  const float* fc3_W  = (const float*)d_in[30];
  const float* fc3_b  = (const float*)d_in[31];
  float* out = (float*)d_out;

  const int* src  = eidx;
  const int* dstn = eidx + CE;
  const int* a0   = assign;
  const int* a1   = assign + CA;
  const int* e20  = eidx2;
  const int* e21  = eidx2 + CE2;

  // ---- workspace layout ----
  char* w = (char*)d_ws;
  size_t off = 0;
  auto alloc = [&](size_t bytes) -> char* { char* p = w + off; off = aln256(off + bytes); return p; };

  __bf16* x_bf    = (__bf16*)alloc((size_t)CN * 64 * 2);
  __bf16* ea_bf   = (__bf16*)alloc((size_t)CE * 32 * 2);
  __bf16* hhid    = (__bf16*)alloc((size_t)CE * 128 * 2);
  __bf16* h0_bf   = (__bf16*)alloc((size_t)CN * 32 * 2);
  float*  h1_f    = (float*) alloc((size_t)CN * 64 * 4);
  float*  agg0    = (float*) alloc((size_t)CN * 32 * 4);
  float*  agg1    = (float*) alloc((size_t)CN * 64 * 4);
  __bf16* Bpk0    = (__bf16*)alloc((size_t)32 * KF0 * 2);
  __bf16* Bpk1    = (__bf16*)alloc((size_t)64 * KF1 * 2);
  __bf16* W1pk0   = (__bf16*)alloc((size_t)128 * 32 * 2);
  __bf16* W1pk1   = (__bf16*)alloc((size_t)128 * 32 * 2);
  __bf16* root0pk = (__bf16*)alloc((size_t)32 * 64 * 2);
  __bf16* root1pk = (__bf16*)alloc((size_t)64 * 32 * 2);
  __bf16* rel4pk  = (__bf16*)alloc((size_t)64 * 96 * 2);
  __bf16* roo4pk  = (__bf16*)alloc((size_t)64 * 96 * 2);
  __bf16* rel5pk  = (__bf16*)alloc((size_t)64 * 64 * 2);
  __bf16* roo5pk  = (__bf16*)alloc((size_t)64 * 64 * 2);
  float*  sum2    = (float*) alloc((size_t)CN2 * 64 * 4);
  float*  cnt2    = (float*) alloc((size_t)CN2 * 4);
  float*  h2_f    = (float*) alloc((size_t)CN2 * 80 * 4);
  __bf16* h2_bf   = (__bf16*)alloc((size_t)CN2 * 96 * 2);
  float*  agg2    = (float*) alloc((size_t)CN2 * 80 * 4);
  __bf16* agg2_bf = (__bf16*)alloc((size_t)CN2 * 96 * 2);
  float*  tmp     = (float*) alloc((size_t)CN2 * 64 * 4);
  float*  h4_f    = (float*) alloc((size_t)CN2 * 64 * 4);
  __bf16* h4_bf   = (__bf16*)alloc((size_t)CN2 * 64 * 2);
  float*  agg5    = (float*) alloc((size_t)CN2 * 64 * 4);
  __bf16* agg5_bf = (__bf16*)alloc((size_t)CN2 * 64 * 2);
  float*  h5_f    = (float*) alloc((size_t)CN2 * 64 * 4);
  float*  psum1   = (float*) alloc((size_t)CG * 64 * 4);
  float*  pcnt1   = (float*) alloc((size_t)CG * 4);
  float*  psum2   = (float*) alloc((size_t)CG * 64 * 4);
  float*  pcnt2   = (float*) alloc((size_t)CG * 4);
  float*  emb     = (float*) alloc((size_t)CG * 128 * 4);
  if (off > ws_size) return;

  const int TB = 256;

  // ---- convert / pack ----
  k_cvt_pad_bf16<<<gridFor((long long)CN * 64, TB), TB, 0, stream>>>(x, x_bf, CN, CF_IN, 64);
  k_cvt_pad_bf16<<<gridFor((long long)CE * 32, TB), TB, 0, stream>>>(edge_attr, ea_bf, CE, CB_IN, 32);
  k_pack_B<<<gridFor(128 * 32, TB), TB, 0, stream>>>(nn0_W1, W1pk0, CB_IN, CHID, 32);
  k_pack_B<<<gridFor(128 * 32, TB), TB, 0, stream>>>(nn1_W1, W1pk1, CB_IN, CHID, 32);
  k_pack_B<<<gridFor(32 * 64, TB), TB, 0, stream>>>(root0, root0pk, CF_IN, 32, 64);
  k_pack_B<<<gridFor(64 * 32, TB), TB, 0, stream>>>(root1, root1pk, 32, CDIM, 32);
  k_pack_B<<<gridFor(64 * 96, TB), TB, 0, stream>>>(W4rel,  rel4pk, 80, CDIM, 96);
  k_pack_B<<<gridFor(64 * 96, TB), TB, 0, stream>>>(W4root, roo4pk, 80, CDIM, 96);
  k_pack_B<<<gridFor(64 * 64, TB), TB, 0, stream>>>(W5rel,  rel5pk, CDIM, CDIM, 64);
  k_pack_B<<<gridFor(64 * 64, TB), TB, 0, stream>>>(W5root, roo5pk, CDIM, CDIM, 64);
  k_pack_W2<<<gridFor((long long)32 * KF0, TB), TB, 0, stream>>>(nn0_W2, nn0_b2, Bpk0, CF_IN, 32, KF0);
  k_pack_W2<<<gridFor((long long)64 * KF1, TB), TB, 0, stream>>>(nn1_W2, nn1_b2, Bpk1, 32, CDIM, KF1);

  // ---- NNConv layer 0 ----
  k_zero_f32<<<gridFor((long long)CN * 32, TB), TB, 0, stream>>>(agg0, (long long)CN * 32);
  // hhid = relu(ea @ W1 + b1)   [E,128] bf16
  k_gemm_wmma<8><<<gridFor(CE, 16), 32, 0, stream>>>(
      ea_bf, 32, W1pk0, 32, nn0_b1, nullptr, nullptr, hhid, 128, CE, 128, 32, 1);
  // fused per-edge dynamic-weight message + scatter
  k_edge_fused<4, 2><<<gridFor(CE, 64), 32, 0, stream>>>(
      x_bf, 64, CF_IN, hhid, Bpk0, KF0, src, dstn, agg0, CE, KF0);
  // h0 = relu(agg0 + x @ root0 + bias0) -> bf16 [N,32]
  k_gemm_wmma<2><<<gridFor(CN, 16), 32, 0, stream>>>(
      x_bf, 64, root0pk, 64, bias0, agg0, nullptr, h0_bf, 32, CN, 32, 64, 1);

  // ---- NNConv layer 1 ----
  k_zero_f32<<<gridFor((long long)CN * 64, TB), TB, 0, stream>>>(agg1, (long long)CN * 64);
  k_gemm_wmma<8><<<gridFor(CE, 16), 32, 0, stream>>>(
      ea_bf, 32, W1pk1, 32, nn1_b1, nullptr, nullptr, hhid, 128, CE, 128, 32, 1);
  k_edge_fused<2, 4><<<gridFor(CE, 32), 32, 0, stream>>>(
      h0_bf, 32, 32, hhid, Bpk1, KF1, src, dstn, agg1, CE, KF1);
  // h1 = relu(agg1 + h0 @ root1 + bias1) -> f32 [N,64]
  k_gemm_wmma<4><<<gridFor(CN, 16), 32, 0, stream>>>(
      h0_bf, 32, root1pk, 32, bias1, agg1, h1_f, nullptr, 0, CN, 64, 32, 1);

  // ---- pool level 1: x_1 = segmean(h1, batch) -> emb[:, :64] ----
  k_zero_f32<<<gridFor(CG * 64, TB), TB, 0, stream>>>(psum1, CG * 64);
  k_zero_f32<<<gridFor(CG, TB), TB, 0, stream>>>(pcnt1, CG);
  k_scatter_add<<<gridFor((long long)CN * 64, TB), TB, 0, stream>>>(
      h1_f, nullptr, batch, psum1, pcnt1, CN, 64);
  k_finalize_emb<<<gridFor(CG * 64, TB), TB, 0, stream>>>(psum1, pcnt1, emb, 0);

  // ---- k-GNN lift: h2 = [segmean(h1[a0], a1) | iso] ----
  k_zero_f32<<<gridFor((long long)CN2 * 64, TB), TB, 0, stream>>>(sum2, (long long)CN2 * 64);
  k_zero_f32<<<gridFor(CN2, TB), TB, 0, stream>>>(cnt2, CN2);
  k_scatter_add<<<gridFor((long long)CA * 64, TB), TB, 0, stream>>>(
      h1_f, a0, a1, sum2, cnt2, CA, 64);
  k_finalize_h2<<<gridFor((long long)CN2 * 96, TB), TB, 0, stream>>>(sum2, cnt2, iso2, h2_f, h2_bf);

  // ---- GraphConv 4: relu(agg2 @ Wrel + h2 @ Wroot + b) ----
  k_zero_f32<<<gridFor((long long)CN2 * 80, TB), TB, 0, stream>>>(agg2, (long long)CN2 * 80);
  k_scatter_add<<<gridFor((long long)CE2 * 80, TB), TB, 0, stream>>>(
      h2_f, e20, e21, agg2, nullptr, CE2, 80);
  k_cvt_pad_bf16<<<gridFor((long long)CN2 * 96, TB), TB, 0, stream>>>(agg2, agg2_bf, CN2, 80, 96);
  k_gemm_wmma<4><<<gridFor(CN2, 16), 32, 0, stream>>>(
      h2_bf, 96, roo4pk, 96, nullptr, nullptr, tmp, nullptr, 0, CN2, 64, 96, 0);
  k_gemm_wmma<4><<<gridFor(CN2, 16), 32, 0, stream>>>(
      agg2_bf, 96, rel4pk, 96, b4, tmp, h4_f, h4_bf, 64, CN2, 64, 96, 1);

  // ---- GraphConv 5 ----
  k_zero_f32<<<gridFor((long long)CN2 * 64, TB), TB, 0, stream>>>(agg5, (long long)CN2 * 64);
  k_scatter_add<<<gridFor((long long)CE2 * 64, TB), TB, 0, stream>>>(
      h4_f, e20, e21, agg5, nullptr, CE2, 64);
  k_cvt_pad_bf16<<<gridFor((long long)CN2 * 64, TB), TB, 0, stream>>>(agg5, agg5_bf, CN2, 64, 64);
  k_gemm_wmma<4><<<gridFor(CN2, 16), 32, 0, stream>>>(
      h4_bf, 64, roo5pk, 64, nullptr, nullptr, tmp, nullptr, 0, CN2, 64, 64, 0);
  k_gemm_wmma<4><<<gridFor(CN2, 16), 32, 0, stream>>>(
      agg5_bf, 64, rel5pk, 64, b5, tmp, h5_f, nullptr, 0, CN2, 64, 64, 1);

  // ---- pool level 2: x_2 = segmean(h5, batch2) -> emb[:, 64:] ----
  k_zero_f32<<<gridFor(CG * 64, TB), TB, 0, stream>>>(psum2, CG * 64);
  k_zero_f32<<<gridFor(CG, TB), TB, 0, stream>>>(pcnt2, CG);
  k_scatter_add<<<gridFor((long long)CN2 * 64, TB), TB, 0, stream>>>(
      h5_f, nullptr, batch2, psum2, pcnt2, CN2, 64);
  k_finalize_emb<<<gridFor(CG * 64, TB), TB, 0, stream>>>(psum2, pcnt2, emb, 64);

  // ---- output MLP ----
  k_final_mlp<<<gridFor(CG, 64), 64, 0, stream>>>(emb, fc1_W, fc1_b, fc2_W, fc2_b,
                                                  fc3_W, fc3_b, out);
}
